// ProjectedAttentionLayer_25632364823137
// MI455X (gfx1250) — matrix operand order repaired and checked by reading, not compiled, not run on hardware
//
#include <hip/hip_runtime.h>
#include <hip/hip_bf16.h>

typedef __bf16 bf16_t;
typedef __attribute__((ext_vector_type(16))) __bf16 v16bf;
typedef __attribute__((ext_vector_type(8)))  __bf16 v8bf;
typedef __attribute__((ext_vector_type(8)))  float  v8f;

constexpr int kB  = 8;
constexpr int kS  = 64;
constexpr int kL  = 512;
constexpr int kD  = 256;
constexpr int kH  = 8;
constexpr int kP  = 32;
constexpr int kHD = 32;
constexpr int kFF = 1024;
constexpr float kEPS = 1e-5f;
constexpr float kScale = 0.17677669529663687f; // 1/sqrt(32)

// ---------------------------------------------------------------------------
// WMMA helpers (CDNA5 16x16x32 bf16, f32 accumulate)
// ---------------------------------------------------------------------------
__device__ __forceinline__ v8f wmma_bf16(v16bf a, v16bf b, v8f c) {
  // 8 args: (neg_a, A, neg_b, B, c_mod, C, reuse_a, reuse_b)
  return __builtin_amdgcn_wmma_f32_16x16x32_bf16(false, a, false, b, (short)0, c,
                                                 false, false);
}

// A fragment (16x32, M x K), row-major source with leading dim `ld`.
// Layout: lanes 0-15 hold row m=lane, K 0..15;
// lanes 16-31 hold the same rows, K 16..31.
__device__ __forceinline__ v16bf frag_a(const bf16_t* base, int ld, int lane) {
  const bf16_t* p = base + (size_t)(lane & 15) * (size_t)ld + ((lane >> 4) << 4);
  v8bf lo = *(const v8bf*)p;         // K = kb .. kb+7
  v8bf hi = *(const v8bf*)(p + 8);   // K = kb+8 .. kb+15
  v16bf a;
#pragma unroll
  for (int i = 0; i < 8; ++i) { a[i] = lo[i]; a[i + 8] = hi[i]; }
  return a;
}

// B fragment (32x16, K x N) where the source stores B *columns* as rows
// (i.e. weight matrix W with W[n][k] = B[k][n]), leading dim `ld`.
// Lanes 0-15: column n=lane, K 0..15; lanes 16-31: same columns, K 16..31.
__device__ __forceinline__ v16bf frag_b(const bf16_t* base, int ld, int lane) {
  const bf16_t* p = base + (size_t)(lane & 15) * (size_t)ld + ((lane >> 4) << 4);
  v8bf lo = *(const v8bf*)p;       // K = kb .. kb+7
  v8bf hi = *(const v8bf*)(p + 8); // K = kb+8 .. kb+15
  v16bf b;
#pragma unroll
  for (int i = 0; i < 8; ++i) { b[i] = lo[i]; b[i + 8] = hi[i]; }
  return b;
}

// B fragment from a genuinely row-major K x N source (strided gather).
__device__ __forceinline__ v16bf frag_b_strided(const bf16_t* base, int ld, int lane) {
  const bf16_t* p = base + (size_t)((lane >> 4) << 4) * (size_t)ld + (lane & 15);
  v16bf b;
#pragma unroll
  for (int i = 0; i < 16; ++i) b[i] = p[(size_t)i * (size_t)ld];
  return b;
}

__device__ __forceinline__ float gelu_exact(float x) {
  return 0.5f * x * (1.0f + erff(x * 0.70710678118654752f));
}

// ---------------------------------------------------------------------------
// f32 -> bf16 cast
// ---------------------------------------------------------------------------
__global__ __launch_bounds__(256) void cast_f32_bf16_kernel(
    const float* __restrict__ src, bf16_t* __restrict__ dst, int n) {
  int i = blockIdx.x * 256 + threadIdx.x;
  if (i < n) dst[i] = (bf16_t)src[i];
}

// ---------------------------------------------------------------------------
// Generic GEMM: Y[N,M] = A[N,K] @ W[M,K]^T + bias, bf16 in/out, f32 accum.
// Register-blocked 16x64 slab per wave (1 A-frag feeds 4 WMMAs) and
// software-pipelined: iteration k+1's fragments are in flight while
// iteration k's WMMAs issue, so waits become partial instead of loadcnt==0.
// Requires M % 64 == 0 (always 256 here).
// ---------------------------------------------------------------------------
__global__ __launch_bounds__(256) void gemm_bf16_kernel(
    const bf16_t* __restrict__ A, const bf16_t* __restrict__ W,
    const float* __restrict__ bias, bf16_t* __restrict__ Y,
    int N, int K, int M) {
  int wid  = blockIdx.x * 8 + (threadIdx.x >> 5);
  int lane = threadIdx.x & 31;
  int cgroups = M >> 6;                 // 64-wide column groups
  int total   = (N >> 4) * cgroups;
  if (wid >= total) return;
  int cg = wid % cgroups;               // consecutive waves share a row tile
  int tn = wid / cgroups;
  size_t row0 = (size_t)tn << 4;
  int col0 = cg << 6;
  const bf16_t* Abase = A + row0 * (size_t)K;
  const bf16_t* Wb    = W + (size_t)col0 * (size_t)K;
  const size_t wstep  = (size_t)16 * (size_t)K;

  v8f acc0 = {}, acc1 = {}, acc2 = {}, acc3 = {};

  // prologue
  v16bf a  = frag_a(Abase, K, lane);
  v16bf b0 = frag_b(Wb, K, lane);
  v16bf b1 = frag_b(Wb + wstep, K, lane);
  v16bf b2 = frag_b(Wb + 2 * wstep, K, lane);
  v16bf b3 = frag_b(Wb + 3 * wstep, K, lane);

  for (int k0 = 32; k0 < K; k0 += 32) {
    // issue next iteration's loads before consuming current fragments
    v16bf an  = frag_a(Abase + k0, K, lane);
    v16bf bn0 = frag_b(Wb + k0, K, lane);
    v16bf bn1 = frag_b(Wb + wstep + k0, K, lane);
    v16bf bn2 = frag_b(Wb + 2 * wstep + k0, K, lane);
    v16bf bn3 = frag_b(Wb + 3 * wstep + k0, K, lane);
    if (k0 + 32 < K)  // prefetch the streamed A operand (global_prefetch_b8)
      __builtin_prefetch(Abase + (size_t)(lane & 15) * (size_t)K + k0 + 32, 0, 1);
    acc0 = wmma_bf16(a, b0, acc0);
    acc1 = wmma_bf16(a, b1, acc1);
    acc2 = wmma_bf16(a, b2, acc2);
    acc3 = wmma_bf16(a, b3, acc3);
    a = an; b0 = bn0; b1 = bn1; b2 = bn2; b3 = bn3;
  }
  // epilogue
  acc0 = wmma_bf16(a, b0, acc0);
  acc1 = wmma_bf16(a, b1, acc1);
  acc2 = wmma_bf16(a, b2, acc2);
  acc3 = wmma_bf16(a, b3, acc3);

  int n  = lane & 15;
  int rb = (lane >> 4) << 3;
#pragma unroll
  for (int j = 0; j < 4; ++j) {
    v8f acc = (j == 0) ? acc0 : (j == 1) ? acc1 : (j == 2) ? acc2 : acc3;
    int c = col0 + j * 16 + n;
    float bv = bias[c];
#pragma unroll
    for (int r = 0; r < 8; ++r)
      Y[(row0 + rb + r) * (size_t)M + c] = (bf16_t)(acc[r] + bv);
  }
}

// ---------------------------------------------------------------------------
// out_attn: one workgroup per (b,s,h). Q: 32xHD (batch-independent),
// K/V: 512xHD. scores 32x512 -> softmax over L -> out 32xHD.
// ---------------------------------------------------------------------------
__global__ __launch_bounds__(256) void attn_out_kernel(
    const bf16_t* __restrict__ Qo, const bf16_t* __restrict__ Ko,
    const bf16_t* __restrict__ Vo, bf16_t* __restrict__ ao) {
  __shared__ __align__(16) bf16_t sc[kP * kL];  // scores, then probs in-place
  __shared__ float red[kP * 8];
  __shared__ float rowmax[kP];
  __shared__ float rowsum[kP];

  int t = threadIdx.x, wave = t >> 5, lane = t & 31;
  int wg = blockIdx.x;        // ((b*S + s)*H + h)
  int h  = wg & (kH - 1);
  int bs = wg >> 3;           // b*S + s
  int s  = bs & (kS - 1);
  const bf16_t* Qb = Qo + (size_t)s * kP * kD + h * kHD;
  const bf16_t* Kb = Ko + (size_t)bs * kL * kD + h * kHD;
  const bf16_t* Vb = Vo + (size_t)bs * kL * kD + h * kHD;

  // scores = Q @ K^T * scale : 2 (P) x 32 (L) tiles, K=HD=32 (single step)
  for (int tt = wave; tt < 64; tt += 8) {
    int tr = tt & 1;    // P tile
    int tc = tt >> 1;   // L tile
    v16bf a = frag_a(Qb + (size_t)(tr * 16) * kD, kD, lane);
    v16bf b = frag_b(Kb + (size_t)(tc * 16) * kD, kD, lane); // B col n = key row
    v8f acc = {};
    acc = wmma_bf16(a, b, acc);
    int n = lane & 15, rb = (lane >> 4) << 3;
#pragma unroll
    for (int r = 0; r < 8; ++r)
      sc[(tr * 16 + rb + r) * kL + tc * 16 + n] = (bf16_t)(acc[r] * kScale);
  }
  __syncthreads();

  // softmax over L=512: 8 threads per row
  {
    int row = t >> 3, j = t & 7, c0 = j * 64;
    float m = -3.0e38f;
    for (int c = 0; c < 64; ++c) m = fmaxf(m, (float)sc[row * kL + c0 + c]);
    red[row * 8 + j] = m;
    __syncthreads();
    if (j == 0) {
      float mm = red[row * 8];
      for (int q = 1; q < 8; ++q) mm = fmaxf(mm, red[row * 8 + q]);
      rowmax[row] = mm;
    }
    __syncthreads();
    float mm = rowmax[row];
    float ssum = 0.f;
    for (int c = 0; c < 64; ++c) ssum += expf((float)sc[row * kL + c0 + c] - mm);
    red[row * 8 + j] = ssum;
    __syncthreads();
    if (j == 0) {
      float acc2 = 0.f;
      for (int q = 0; q < 8; ++q) acc2 += red[row * 8 + q];
      rowsum[row] = acc2;
    }
    __syncthreads();
    float inv = 1.0f / rowsum[row];
    for (int c = 0; c < 64; ++c) {
      int idx = row * kL + c0 + c;
      sc[idx] = (bf16_t)(expf((float)sc[idx] - mm) * inv);
    }
  }
  __syncthreads();

  // out = probs(32x512) @ V(512xHD): 2x2 tiles, K=512
  if (wave < 4) {
    int tr = wave >> 1, tm2 = wave & 1;
    v8f acc = {};
    for (int k0 = 0; k0 < kL; k0 += 32) {
      v16bf a = frag_a(&sc[(tr * 16) * kL + k0], kL, lane);
      v16bf b = frag_b_strided(Vb + (size_t)k0 * kD + tm2 * 16, kD, lane);
      acc = wmma_bf16(a, b, acc);
    }
    int n = lane & 15, rb = (lane >> 4) << 3;
    bf16_t* ob = ao + (size_t)bs * kP * kD + h * kHD;
#pragma unroll
    for (int r = 0; r < 8; ++r)
      ob[(size_t)(tr * 16 + rb + r) * kD + tm2 * 16 + n] = (bf16_t)acc[r];
  }
}

// ---------------------------------------------------------------------------
// in_attn: one workgroup per (b,s,h). Q: 512xHD, K: 32xHD (batch-indep),
// V: 32xHD. scores 512x32 -> softmax over P=32 -> out 512xHD.
// ---------------------------------------------------------------------------
__global__ __launch_bounds__(256) void attn_in_kernel(
    const bf16_t* __restrict__ Qi, const bf16_t* __restrict__ Ki,
    const bf16_t* __restrict__ Vi, bf16_t* __restrict__ ai) {
  __shared__ __align__(16) bf16_t sc[kL * kP];   // 512x32 scores -> probs
  __shared__ __align__(16) bf16_t Vt[kHD * kP];  // V^T (HD x P)

  int t = threadIdx.x, wave = t >> 5, lane = t & 31;
  int wg = blockIdx.x;
  int h  = wg & (kH - 1);
  int bs = wg >> 3;
  int s  = bs & (kS - 1);
  const bf16_t* Qb = Qi + (size_t)bs * kL * kD + h * kHD;
  const bf16_t* Kb = Ki + (size_t)s * kP * kD + h * kHD;
  const bf16_t* Vb = Vi + (size_t)bs * kP * kD + h * kHD;

  // stage V^T (32x32)
  for (int i = t; i < kP * kHD; i += 256) {
    int p = i >> 5, c = i & 31;
    Vt[c * kP + p] = Vb[(size_t)p * kD + c];
  }

  // scores = Q @ K^T * scale : 32 (L) x 2 (P) tiles, K=HD=32
  for (int tt = wave; tt < 64; tt += 8) {
    int tc = tt & 1;    // P tile
    int tr = tt >> 1;   // L tile
    v16bf a = frag_a(Qb + (size_t)(tr * 16) * kD, kD, lane);
    v16bf b = frag_b(Kb + (size_t)(tc * 16) * kD, kD, lane);
    v8f acc = {};
    acc = wmma_bf16(a, b, acc);
    int n = lane & 15, rb = (lane >> 4) << 3;
#pragma unroll
    for (int r = 0; r < 8; ++r)
      sc[(tr * 16 + rb + r) * kP + tc * 16 + n] = (bf16_t)(acc[r] * kScale);
  }
  __syncthreads();

  // softmax over P=32, two rows per thread (rows owned exclusively)
#pragma unroll
  for (int rr = 0; rr < 2; ++rr) {
    int row = t * 2 + rr;
    float m = -3.0e38f;
    for (int c = 0; c < kP; ++c) m = fmaxf(m, (float)sc[row * kP + c]);
    float ssum = 0.f;
    for (int c = 0; c < kP; ++c) ssum += expf((float)sc[row * kP + c] - m);
    float inv = 1.0f / ssum;
    for (int c = 0; c < kP; ++c)
      sc[row * kP + c] = (bf16_t)(expf((float)sc[row * kP + c] - m) * inv);
  }
  __syncthreads();

  // out = probs(512x32) @ V(32xHD): 32x2 tiles, K=32
  for (int tt = wave; tt < 64; tt += 8) {
    int tm2 = tt & 1;
    int tr  = tt >> 1;
    v16bf a = frag_a(&sc[(tr * 16) * kP], kP, lane);
    v16bf b = frag_b(&Vt[(tm2 * 16) * kP], kP, lane);
    v8f acc = {};
    acc = wmma_bf16(a, b, acc);
    int n = lane & 15, rb = (lane >> 4) << 3;
    bf16_t* ob = ai + (size_t)bs * kL * kD + h * kHD;
#pragma unroll
    for (int r = 0; r < 8; ++r)
      ob[(size_t)(tr * 16 + rb + r) * kD + tm2 * 16 + n] = (bf16_t)acc[r];
  }
}

// ---------------------------------------------------------------------------
// LN1: m = LayerNorm(x + message_in); one block (256 thr) per row of 256
// ---------------------------------------------------------------------------
__global__ __launch_bounds__(256) void ln1_kernel(
    const float* __restrict__ x, const bf16_t* __restrict__ mi,
    const float* __restrict__ g, const float* __restrict__ b,
    float* __restrict__ m32, bf16_t* __restrict__ mb) {
  __shared__ float red[256];
  int t = threadIdx.x;
  size_t idx = (size_t)blockIdx.x * kD + t;
  float v = x[idx] + (float)mi[idx];
  red[t] = v;
  __syncthreads();
  for (int s2 = 128; s2 > 0; s2 >>= 1) {
    if (t < s2) red[t] += red[t + s2];
    __syncthreads();
  }
  float mu = red[0] * (1.0f / kD);
  __syncthreads();
  float d = v - mu;
  red[t] = d * d;
  __syncthreads();
  for (int s2 = 128; s2 > 0; s2 >>= 1) {
    if (t < s2) red[t] += red[t + s2];
    __syncthreads();
  }
  float var = red[0] * (1.0f / kD);
  float y = d * rsqrtf(var + kEPS) * g[t] + b[t];
  m32[idx] = y;
  mb[idx]  = (bf16_t)y;
}

// ---------------------------------------------------------------------------
// Fused MLP + LN2: per block, 16 rows: h1 = gelu(m@W1^T+b1) (kept in LDS),
// h = h1@W2^T+b2, out = LN(m + h). Never materializes h1/h in HBM.
// ---------------------------------------------------------------------------
__global__ __launch_bounds__(256) void mlp_ln2_kernel(
    const bf16_t* __restrict__ mb, const float* __restrict__ m32,
    const bf16_t* __restrict__ w1, const float* __restrict__ b1,
    const bf16_t* __restrict__ w2, const float* __restrict__ b2,
    const float* __restrict__ g2, const float* __restrict__ bb2,
    float* __restrict__ out) {
  __shared__ __align__(16) bf16_t mrow[16 * kD];   // 8 KB
  __shared__ __align__(16) bf16_t h1[16 * kFF];    // 32 KB
  __shared__ float ht[16 * kD];                    // 16 KB
  __shared__ float redS[256];
  __shared__ float redQ[256];

  int t = threadIdx.x, wave = t >> 5, lane = t & 31;
  size_t r0 = (size_t)blockIdx.x * 16;

  for (int i = t; i < 16 * kD; i += 256) mrow[i] = mb[r0 * kD + i];
  __syncthreads();

  // h1 = gelu(m @ W1^T + b1): 64 column tiles, K=256.
  // The 8 A-fragments are invariant across tiles -> hoist into registers.
  {
    v16bf afr[8];
#pragma unroll
    for (int kk = 0; kk < 8; ++kk) afr[kk] = frag_a(&mrow[kk * 32], kD, lane);
    for (int tt = wave; tt < kFF / 16; tt += 8) {
      int col0 = tt * 16;
      const bf16_t* wb = w1 + (size_t)col0 * kD;
      v8f acc = {};
#pragma unroll
      for (int kk = 0; kk < 8; ++kk) {
        v16bf b = frag_b(wb + kk * 32, kD, lane);
        acc = wmma_bf16(afr[kk], b, acc);
      }
      int n = lane & 15, rb = (lane >> 4) << 3;
      float bv = b1[col0 + n];
#pragma unroll
      for (int r = 0; r < 8; ++r)
        h1[(rb + r) * kFF + col0 + n] = (bf16_t)gelu_exact(acc[r] + bv);
    }
  }
  __syncthreads();

  // h = h1 @ W2^T + b2: 16 column tiles, K=1024. Each wave owns 2 tiles,
  // blocked together and software-pipelined on the global W2 stream.
  {
    int c0a = wave * 16, c0b = (wave + 8) * 16;
    const bf16_t* wba = w2 + (size_t)c0a * kFF;
    const bf16_t* wbb = w2 + (size_t)c0b * kFF;
    v8f acc0 = {}, acc1 = {};
    v16bf a  = frag_a(&h1[0], kFF, lane);
    v16bf ba = frag_b(wba, kFF, lane);
    v16bf bb = frag_b(wbb, kFF, lane);
    for (int k0 = 32; k0 < kFF; k0 += 32) {
      v16bf an  = frag_a(&h1[k0], kFF, lane);
      v16bf bna = frag_b(wba + k0, kFF, lane);
      v16bf bnb = frag_b(wbb + k0, kFF, lane);
      acc0 = wmma_bf16(a, ba, acc0);
      acc1 = wmma_bf16(a, bb, acc1);
      a = an; ba = bna; bb = bnb;
    }
    acc0 = wmma_bf16(a, ba, acc0);
    acc1 = wmma_bf16(a, bb, acc1);
    int n = lane & 15, rb = (lane >> 4) << 3;
    float bva = b2[c0a + n], bvb = b2[c0b + n];
#pragma unroll
    for (int r = 0; r < 8; ++r) {
      ht[(rb + r) * kD + c0a + n] = acc0[r] + bva;
      ht[(rb + r) * kD + c0b + n] = acc1[r] + bvb;
    }
  }
  __syncthreads();

  // LN2 over m + h: 16 threads per row, 16 cols each
  int row = t >> 4, c0 = t & 15;
  float vals[16];
  float s = 0.f, q = 0.f;
#pragma unroll
  for (int j = 0; j < 16; ++j) {
    int c = c0 + j * 16;
    float v = m32[(r0 + row) * kD + c] + ht[row * kD + c];
    vals[j] = v;
    s += v;
    q += v * v;
  }
  redS[t] = s;
  redQ[t] = q;
  __syncthreads();
  if (c0 == 0) {
    float S = 0.f, Q = 0.f;
    for (int j = 0; j < 16; ++j) { S += redS[row * 16 + j]; Q += redQ[row * 16 + j]; }
    redS[row * 16] = S;
    redQ[row * 16] = Q;
  }
  __syncthreads();
  float mu  = redS[row * 16] * (1.0f / kD);
  float var = redQ[row * 16] * (1.0f / kD) - mu * mu;
  float inv = rsqrtf(var + kEPS);
#pragma unroll
  for (int j = 0; j < 16; ++j) {
    int c = c0 + j * 16;
    out[(r0 + row) * kD + c] = (vals[j] - mu) * inv * g2[c] + bb2[c];
  }
}

// ---------------------------------------------------------------------------
// Host orchestration
// ---------------------------------------------------------------------------
extern "C" void kernel_launch(void* const* d_in, const int* in_sizes, int n_in,
                              void* d_out, int out_size, void* d_ws, size_t ws_size,
                              hipStream_t stream) {
  (void)in_sizes; (void)n_in; (void)out_size; (void)ws_size;

  const float* x     = (const float*)d_in[0];
  const float* prj   = (const float*)d_in[1];
  const float* Wq_o  = (const float*)d_in[2];  const float* bq_o = (const float*)d_in[3];
  const float* Wk_o  = (const float*)d_in[4];  const float* bk_o = (const float*)d_in[5];
  const float* Wv_o  = (const float*)d_in[6];  const float* bv_o = (const float*)d_in[7];
  const float* Wo_o  = (const float*)d_in[8];  const float* bo_o = (const float*)d_in[9];
  const float* Wq_i  = (const float*)d_in[10]; const float* bq_i = (const float*)d_in[11];
  const float* Wk_i  = (const float*)d_in[12]; const float* bk_i = (const float*)d_in[13];
  const float* Wv_i  = (const float*)d_in[14]; const float* bv_i = (const float*)d_in[15];
  const float* Wo_i  = (const float*)d_in[16]; const float* bo_i = (const float*)d_in[17];
  const float* g1    = (const float*)d_in[18]; const float* b1   = (const float*)d_in[19];
  const float* g2    = (const float*)d_in[20]; const float* b2   = (const float*)d_in[21];
  const float* W_ff1 = (const float*)d_in[22]; const float* b_ff1 = (const float*)d_in[23];
  const float* W_ff2 = (const float*)d_in[24]; const float* b_ff2 = (const float*)d_in[25];
  float* out = (float*)d_out;

  const int NX   = kB * kS * kL * kD;   // 67,108,864
  const int NTOK = kB * kS * kL;        // 262,144
  const int NP   = kS * kP;             // 2,048  (batch-independent tokens)
  const int NBP  = kB * kS * kP;        // 16,384
  const int WDD  = kD * kD;             // 65,536
  const int WFD  = kFF * kD;            // 262,144

  char* wsp = (char*)d_ws;
  size_t off = 0;
  auto take = [&](size_t bytes) -> void* {
    void* p = wsp + off;
    off += (bytes + 255) & ~(size_t)255;
    return p;
  };

  bf16_t* xb  = (bf16_t*)take((size_t)NX * 2);
  bf16_t* pb  = (bf16_t*)take((size_t)NP * kD * 2);
  bf16_t* wqo = (bf16_t*)take((size_t)WDD * 2);
  bf16_t* wko = (bf16_t*)take((size_t)WDD * 2);
  bf16_t* wvo = (bf16_t*)take((size_t)WDD * 2);
  bf16_t* woo = (bf16_t*)take((size_t)WDD * 2);
  bf16_t* wqi = (bf16_t*)take((size_t)WDD * 2);
  bf16_t* wki = (bf16_t*)take((size_t)WDD * 2);
  bf16_t* wvi = (bf16_t*)take((size_t)WDD * 2);
  bf16_t* woi = (bf16_t*)take((size_t)WDD * 2);
  bf16_t* wf1 = (bf16_t*)take((size_t)WFD * 2);
  bf16_t* wf2 = (bf16_t*)take((size_t)WFD * 2);
  bf16_t* Qo  = (bf16_t*)take((size_t)NP * kD * 2);
  bf16_t* Kib = (bf16_t*)take((size_t)NP * kD * 2);
  bf16_t* Ko  = (bf16_t*)take((size_t)NX * 2);
  bf16_t* Vo  = (bf16_t*)take((size_t)NX * 2);
  bf16_t* ao  = (bf16_t*)take((size_t)NBP * kD * 2);
  bf16_t* mo  = (bf16_t*)take((size_t)NBP * kD * 2);
  bf16_t* Vi  = (bf16_t*)take((size_t)NBP * kD * 2);
  bf16_t* Qi  = (bf16_t*)take((size_t)NX * 2);
  float*  m32 = (float*)take((size_t)NX * 4);
  // Aliased (lifetimes disjoint):
  bf16_t* ai = Ko;  // in_attn output reuses K_out space
  bf16_t* mi = Vo;  // message_in reuses V_out space
  bf16_t* mb = Qi;  // bf16 LN1 output reuses Q_in space

  auto cast = [&](const float* s, bf16_t* d, int n) {
    cast_f32_bf16_kernel<<<(n + 255) / 256, 256, 0, stream>>>(s, d, n);
  };
  cast(x, xb, NX);
  cast(prj, pb, NP * kD);
  cast(Wq_o, wqo, WDD); cast(Wk_o, wko, WDD);
  cast(Wv_o, wvo, WDD); cast(Wo_o, woo, WDD);
  cast(Wq_i, wqi, WDD); cast(Wk_i, wki, WDD);
  cast(Wv_i, wvi, WDD); cast(Wo_i, woi, WDD);
  cast(W_ff1, wf1, WFD); cast(W_ff2, wf2, WFD);

  auto gemm = [&](const bf16_t* A, const bf16_t* W, const float* bias, bf16_t* Y,
                  int N, int K, int M) {
    int tiles = (N / 16) * (M / 64);   // 16x64 slab per wave
    gemm_bf16_kernel<<<(tiles + 7) / 8, 256, 0, stream>>>(A, W, bias, Y, N, K, M);
  };

  // out_attn: projections (Q computed once over S*P, K/V over all tokens)
  gemm(xb, wko, bk_o, Ko, NTOK, kD, kD);
  gemm(xb, wvo, bv_o, Vo, NTOK, kD, kD);
  gemm(pb, wqo, bq_o, Qo, NP, kD, kD);
  attn_out_kernel<<<kB * kS * kH, 256, 0, stream>>>(Qo, Ko, Vo, ao);
  gemm(ao, woo, bo_o, mo, NBP, kD, kD);   // message_out

  // in_attn
  gemm(pb, wki, bk_i, Kib, NP, kD, kD);   // K_in (batch-independent)
  gemm(mo, wvi, bv_i, Vi, NBP, kD, kD);
  gemm(xb, wqi, bq_i, Qi, NTOK, kD, kD);
  attn_in_kernel<<<kB * kS * kH, 256, 0, stream>>>(Qi, Kib, Vi, ai);
  gemm(ai, woi, bo_i, mi, NTOK, kD, kD);  // message_in

  // m = LN(x + message_in)
  ln1_kernel<<<NTOK, 256, 0, stream>>>(x, mi, g1, b1, m32, mb);

  // out = LN(m + MLP(m)) fully fused per 16 rows
  mlp_ln2_kernel<<<NTOK / 16, 256, 0, stream>>>(mb, m32, wf1, b_ff1, wf2, b_ff2,
                                                g2, b2, out);
}
